// AudioQuantizer_87754771792646
// MI455X (gfx1250) — matrix operand order repaired and checked by reading, not compiled, not run on hardware
//
#include <hip/hip_runtime.h>
#include <hip/hip_bf16.h>

#define D_MODEL 512
#define D_PAIRS 256  // packed bf16 pairs (uints) per row

typedef __attribute__((ext_vector_type(16))) __bf16 bf16x16;
typedef __attribute__((ext_vector_type(8)))  float  f32x8;

#if __has_builtin(__builtin_amdgcn_global_load_async_to_lds_b128)
#define HAVE_ASYNC_LDS 1
#else
#define HAVE_ASYNC_LDS 0
#endif

__device__ __forceinline__ unsigned int bf16_rne(unsigned int u) {
  return (u + 0x7FFFu + ((u >> 16) & 1u)) >> 16;
}
__device__ __forceinline__ unsigned int pack2(float a, float b) {
  return bf16_rne(__float_as_uint(a)) | (bf16_rne(__float_as_uint(b)) << 16);
}

#if HAVE_ASYNC_LDS
// Builtin prototype: (v4i AS1* gsrc, v4i AS3* ldst, imm offset, imm cpol),
// with v4i a GCC-vector 'int __vector_size__(16)'.
typedef int v4i_gcc __attribute__((vector_size(16)));
typedef __attribute__((address_space(1))) v4i_gcc as1_v4i;
typedef __attribute__((address_space(3))) v4i_gcc as3_v4i;

__device__ __forceinline__ void async_copy16(const unsigned int* g, unsigned int* l) {
  __builtin_amdgcn_global_load_async_to_lds_b128(
      (as1_v4i*)(unsigned long long)g,
      (as3_v4i*)(unsigned int)(unsigned long long)l,
      0, 0);
}
__device__ __forceinline__ void wait_async0() {
#if __has_builtin(__builtin_amdgcn_s_wait_asynccnt)
  __builtin_amdgcn_s_wait_asynccnt(0);
#else
  asm volatile("s_wait_asynccnt 0x0" ::: "memory");
#endif
}
#endif

// Stage 64 bf16 codebook rows (64 KB) into LDS; 256 threads x 16 x b128 DMA.
__device__ __forceinline__ void stage_chunk64(const unsigned int* __restrict__ src,
                                              unsigned int* __restrict__ dst, int tid) {
#pragma unroll
  for (int i = 0; i < 16; ++i) {
#if HAVE_ASYNC_LDS
    async_copy16(src + (tid + i * 256) * 4, dst + (tid + i * 256) * 4);
#else
    ((uint4*)dst)[tid + i * 256] = ((const uint4*)src)[tid + i * 256];
#endif
  }
}

// ---------------------------------------------------------------------------
// Kernel 1: codebook f32 -> packed bf16 (ws) + per-row sum of squares (f32).
// ---------------------------------------------------------------------------
__global__ void vq_prep_codebook(const float* __restrict__ cb,
                                 unsigned int* __restrict__ cbbf,
                                 float* __restrict__ cb_sq) {
  __shared__ float rs[256];
  const int row = blockIdx.x;
  const int tid = threadIdx.x;
  const float2* src = (const float2*)(cb + (size_t)row * D_MODEL);
  float2 p = src[tid];
  cbbf[(size_t)row * D_PAIRS + tid] = pack2(p.x, p.y);
  rs[tid] = p.x * p.x + p.y * p.y;
  __syncthreads();
  for (int off = 128; off > 0; off >>= 1) {
    if (tid < off) rs[tid] += rs[tid + off];
    __syncthreads();
  }
  if (tid == 0) cb_sq[row] = rs[0];
}

// ---------------------------------------------------------------------------
// Kernel 2: 128 tokens/block. A-fragments resident in VGPRs; 64-row B stages
// double-buffered in LDS via async DMA (64 WMMAs per barrier); 3-slot B
// register pipeline; dual WMMA accumulation chains; c_sq folded into the
// accumulator init and prefetched one stage ahead; fused embedding gather.
// ---------------------------------------------------------------------------
__global__ __launch_bounds__(256, 1)
void vq_argmin_gather(const float* __restrict__ x,
                      const unsigned int* __restrict__ cbbf,
                      const float* __restrict__ cb_sq,
                      const float* __restrict__ emb,
                      float* __restrict__ out,
                      int K) {
  __shared__ unsigned int btile[2][64 * D_PAIRS];  // 2 x 64 KB
  __shared__ float redv[8 * 32 * 8];               // 8 KB
  __shared__ int   redi[8 * 32 * 8];               // 8 KB
  __shared__ int   bidx[128];

  const int tid   = threadIdx.x;
  const int wave  = tid >> 5;
  const int lane  = tid & 31;
  const int lhalf = lane >> 4;
  const int lmod  = lane & 15;
  const long long tok0 = (long long)blockIdx.x * 128;

  // ---- Preload this wave's 16 A fragments (16 k-steps x 32 bf16) into regs.
  // A 16x32 bf16 layout: lane<16 -> K {kk..kk+7, kk+16..kk+23},
  //                      lane>=16 -> K {kk+8..kk+15, kk+24..kk+31}
  uint4 areg[32];  // 128 VGPRs
  {
    const float* xr = x + (tok0 + wave * 16 + lmod) * D_MODEL;
#pragma unroll
    for (int ks = 0; ks < 16; ++ks) {
      const int d0 = ks * 32 + lhalf * 8;
      float4 q0 = *(const float4*)(xr + d0);
      float4 q1 = *(const float4*)(xr + d0 + 4);
      float4 q2 = *(const float4*)(xr + d0 + 16);
      float4 q3 = *(const float4*)(xr + d0 + 20);
      areg[2 * ks]     = make_uint4(pack2(q0.x, q0.y), pack2(q0.z, q0.w),
                                    pack2(q1.x, q1.y), pack2(q1.z, q1.w));
      areg[2 * ks + 1] = make_uint4(pack2(q2.x, q2.y), pack2(q2.z, q2.w),
                                    pack2(q3.x, q3.y), pack2(q3.z, q3.w));
    }
  }

  float bestv[8];
  int   besti[8];
#pragma unroll
  for (int j = 0; j < 8; ++j) { bestv[j] = -3.402823466e+38f; besti[j] = 0; }

  // Prologue: stage rows 0..63 into buffer 0; prefetch first c_sq quad.
  stage_chunk64(cbbf, &btile[0][0], tid);
  float cs_next[4];
#pragma unroll
  for (int s = 0; s < 4; ++s) cs_next[s] = cb_sq[s * 16 + lmod];

  int it = 0;
  for (int nb = 0; nb < K; nb += 64, ++it) {
    const int p = it & 1;
    const unsigned int* bufp = &btile[p][0];
    unsigned int* bufn = &btile[p ^ 1][0];

#if HAVE_ASYNC_LDS
    wait_async0();       // my stage-it DMAs complete
#endif
    __syncthreads();     // everyone's DMAs done AND prev compute on bufn done
    if (nb + 64 < K) stage_chunk64(cbbf + (size_t)(nb + 64) * D_PAIRS, bufn, tid);

    float cs_cur[4];
#pragma unroll
    for (int s = 0; s < 4; ++s) cs_cur[s] = cs_next[s];
    if (nb + 64 < K) {
#pragma unroll
      for (int s = 0; s < 4; ++s) cs_next[s] = cb_sq[nb + 64 + s * 16 + lmod];
    }

#pragma unroll
    for (int s = 0; s < 4; ++s) {
      const float c0 = -0.5f * cs_cur[s];
      f32x8 acc0, acc1;
#pragma unroll
      for (int j = 0; j < 8; ++j) { acc0[j] = c0; acc1[j] = 0.0f; }

      // B 32x16 bf16: col = lane%16; lane<16 -> K {kk..kk+7, kk+16..kk+23}, lane>=16 -> +8.
      // Per-step uint offset advances by 16; 3-slot software pipeline.
      const unsigned int* bbase = bufp + (s * 16 + lmod) * D_PAIRS + lhalf * 4;
      uint4 sb0[3], sb1[3];
#pragma unroll
      for (int w = 0; w < 2; ++w) {
        sb0[w] = *(const uint4*)&bbase[w * 16];
        sb1[w] = *(const uint4*)&bbase[w * 16 + 8];
      }
#pragma unroll
      for (int ks = 0; ks < 16; ++ks) {
        const int cur = ks % 3;
        if (ks + 2 < 16) {
          const int nx = (ks + 2) % 3;
          const unsigned int* bn = bbase + (ks + 2) * 16;
          sb0[nx] = *(const uint4*)&bn[0];
          sb1[nx] = *(const uint4*)&bn[8];
        }
        union { uint4 u[2]; bf16x16 v; } fa, fb;
        fa.u[0] = areg[2 * ks]; fa.u[1] = areg[2 * ks + 1];
        fb.u[0] = sb0[cur];     fb.u[1] = sb1[cur];
        if (ks & 1)
          acc1 = __builtin_amdgcn_wmma_f32_16x16x32_bf16(
              false, fa.v, false, fb.v, (short)0, acc1, false, false);
        else
          acc0 = __builtin_amdgcn_wmma_f32_16x16x32_bf16(
              false, fa.v, false, fb.v, (short)0, acc0, false, false);
      }

      // score = dot - c_sq/2; larger == closer. Strict > keeps lowest index.
#pragma unroll
      for (int j = 0; j < 8; ++j) {
        const float sc = acc0[j] + acc1[j];
        if (sc > bestv[j]) { bestv[j] = sc; besti[j] = nb + s * 16 + lmod; }
      }
    }
  }

  // Cross-lane argmax: C element (lane, j) = token j+8*(lane>=16), col lane%16.
#pragma unroll
  for (int j = 0; j < 8; ++j) {
    redv[(wave * 32 + lane) * 8 + j] = bestv[j];
    redi[(wave * 32 + lane) * 8 + j] = besti[j];
  }
  __syncthreads();
  if (lane < 16) {
    const int t  = lane;
    const int sj = t & 7;
    const int sh = t >> 3;
    float bv = -3.402823466e+38f;
    int   bi = 0x7FFFFFFF;
#pragma unroll
    for (int i = 0; i < 16; ++i) {
      float v  = redv[(wave * 32 + sh * 16 + i) * 8 + sj];
      int   ix = redi[(wave * 32 + sh * 16 + i) * 8 + sj];
      if (v > bv || (v == bv && ix < bi)) { bv = v; bi = ix; }
    }
    bidx[wave * 16 + t] = bi;
  }
  __syncthreads();

  // Fused gather: out[token] = embedding[best_idx], float4 copies.
  const float4* emb4 = (const float4*)emb;
  float4* out4 = (float4*)out;
  for (int i = tid; i < 128 * 128; i += 256) {
    const int t  = i >> 7;
    const int d4 = i & 127;
    out4[(tok0 + t) * 128 + d4] = emb4[(size_t)bidx[t] * 128 + d4];
  }
}

extern "C" void kernel_launch(void* const* d_in, const int* in_sizes, int n_in,
                              void* d_out, int out_size, void* d_ws, size_t ws_size,
                              hipStream_t stream) {
  const float* x   = (const float*)d_in[0];
  const float* cb  = (const float*)d_in[1];
  const float* emb = (const float*)d_in[2];
  float* out = (float*)d_out;

  const int D = D_MODEL;
  const int K = in_sizes[1] / D;            // 8192 codes
  const long long T = in_sizes[0] / D;      // 32768 tokens

  // ws layout: [ K*D bf16 codebook (8 MB) | K f32 c_sq (32 KB) ]
  unsigned int* cbbf = (unsigned int*)d_ws;
  float* cb_sq = (float*)((char*)d_ws + (size_t)K * D * 2);

  vq_prep_codebook<<<K, 256, 0, stream>>>(cb, cbbf, cb_sq);
  vq_argmin_gather<<<(int)(T / 128), 256, 0, stream>>>(x, cbbf, cb_sq, emb, out, K);
}
